// MultiHeadAttention_1580547975427
// MI455X (gfx1250) — matrix operand order repaired and checked by reading, not compiled, not run on hardware
//
#include <hip/hip_runtime.h>

// ---------------------------------------------------------------------------
// MHA block for MI455X (gfx1250, wave32, WMMA bf16 16x16x32, f32 accumulate)
// B=8, L=1024, D=1024, H=16, DPH=64  ->  BH=128, M=B*L=8192
// ---------------------------------------------------------------------------

typedef __attribute__((ext_vector_type(16))) __bf16 v16bf;
typedef __attribute__((ext_vector_type(8)))  __bf16 v8bf;
typedef __attribute__((ext_vector_type(8)))  float  v8f;

__device__ __forceinline__ unsigned short f2bf_u16(float f) {
  unsigned int u = __builtin_bit_cast(unsigned int, f);
  u = (u + 0x7FFFu + ((u >> 16) & 1u)) >> 16;   // round-to-nearest-even
  return (unsigned short)u;
}
__device__ __forceinline__ __bf16 f2bf(float f) {
  unsigned short s = f2bf_u16(f);
  return __builtin_bit_cast(__bf16, s);
}
__device__ __forceinline__ v8f wmma_bf16(v16bf a, v16bf b, v8f c) {
  return __builtin_amdgcn_wmma_f32_16x16x32_bf16(false, a, false, b, (short)0, c,
                                                 false, false);
}
// A-fragment (16x32 bf16): two 8-half contiguous chunks at p and p+16 halves.
__device__ __forceinline__ v16bf load_frag_a(const unsigned short* p) {
  v8bf lo = *reinterpret_cast<const v8bf*>(p);
  v8bf hl = *reinterpret_cast<const v8bf*>(p + 16);
  return __builtin_shufflevector(lo, hl, 0, 1, 2, 3, 4, 5, 6, 7, 8, 9, 10, 11,
                                 12, 13, 14, 15);
}
// B-fragment (32x16 bf16): two 8-half contiguous chunks at p and p+16 halves.
__device__ __forceinline__ v16bf load_frag_b(const unsigned short* p) {
  v8bf lo = *reinterpret_cast<const v8bf*>(p);
  v8bf hl = *reinterpret_cast<const v8bf*>(p + 16);
  return __builtin_shufflevector(lo, hl, 0, 1, 2, 3, 4, 5, 6, 7, 8, 9, 10, 11,
                                 12, 13, 14, 15);
}

// ---------------------------------------------------------------------------
__global__ void cvt_bf16(const float* __restrict__ src,
                         unsigned short* __restrict__ dst, int n) {
  int i = blockIdx.x * blockDim.x + threadIdx.x;
  if (i < n) dst[i] = f2bf_u16(src[i]);
}

// Tiled 1024x1024 transpose + convert: Wt[n*1024+k] = bf16(W[k*1024+n]).
// 64x64 tile per block via LDS, coalesced loads and stores.
__global__ __launch_bounds__(256) void transpose_cvt_1024(
    const float* __restrict__ W, unsigned short* __restrict__ Wt) {
  __shared__ float t[64][65];
  const int tx = threadIdx.x & 63;   // fast dim
  const int ty = threadIdx.x >> 6;   // 0..3
  const int n0 = blockIdx.x * 64;
  const int k0 = blockIdx.y * 64;
#pragma unroll
  for (int j = 0; j < 16; ++j) {
    const int k = k0 + ty + j * 4;
    t[ty + j * 4][tx] = W[(size_t)k * 1024 + n0 + tx];
  }
  __syncthreads();
#pragma unroll
  for (int j = 0; j < 16; ++j) {
    const int n = n0 + ty + j * 4;
    Wt[(size_t)n * 1024 + k0 + tx] = f2bf_u16(t[tx][ty + j * 4]);
  }
}

// ---------------------------------------------------------------------------
// WMMA bf16 GEMM: C = A[M,K] @ B + bias, Bt pre-transposed [N,K] bf16.
// 256 thr = 8 waves, block tile 128x128, wave tile 64x32, BK=32,
// software-pipelined fragment loads.
// MODE 0: bf16 row-major. MODE 1: bf16 V-transposed ([bh][dph][key]).
// MODE 2: f32 row-major.
// ---------------------------------------------------------------------------
template <int MODE>
__global__ __launch_bounds__(256) void gemm_bf16_wmma(
    const unsigned short* __restrict__ A,
    const unsigned short* __restrict__ Bt,
    const float* __restrict__ bias,
    void* __restrict__ Cout, int M, int N, int K) {
  const int tid  = threadIdx.x;
  const int wave = tid >> 5;
  const int lane = tid & 31;
  const int l16  = lane & 15;
  const int hi   = lane >> 4;

  const int rm = blockIdx.y * 128 + (wave & 1) * 64;
  const int rn = blockIdx.x * 128 + (wave >> 1) * 32;

  const unsigned short* Abase[4];
#pragma unroll
  for (int mt = 0; mt < 4; ++mt)
    Abase[mt] = A + (size_t)(rm + mt * 16 + l16) * K + hi * 8;
  const unsigned short* Bbase[2];
#pragma unroll
  for (int nt = 0; nt < 2; ++nt)
    Bbase[nt] = Bt + (size_t)(rn + nt * 16 + l16) * K + hi * 8;

  v8f zero = {0.f, 0.f, 0.f, 0.f, 0.f, 0.f, 0.f, 0.f};
  v8f acc[4][2];
#pragma unroll
  for (int mt = 0; mt < 4; ++mt)
#pragma unroll
    for (int nt = 0; nt < 2; ++nt) acc[mt][nt] = zero;

  v16bf a0[4], b0[2];
#pragma unroll
  for (int mt = 0; mt < 4; ++mt) a0[mt] = load_frag_a(Abase[mt]);
#pragma unroll
  for (int nt = 0; nt < 2; ++nt) b0[nt] = load_frag_b(Bbase[nt]);

  for (int kk = 0; kk < K - 32; kk += 32) {
    v16bf a1[4], b1[2];
#pragma unroll
    for (int mt = 0; mt < 4; ++mt) a1[mt] = load_frag_a(Abase[mt] + kk + 32);
#pragma unroll
    for (int nt = 0; nt < 2; ++nt) b1[nt] = load_frag_b(Bbase[nt] + kk + 32);
    if (kk + 96 < K) {
      __builtin_prefetch(Bbase[0] + kk + 96, 0, 3);
      __builtin_prefetch(Bbase[1] + kk + 96, 0, 3);
    }
#pragma unroll
    for (int mt = 0; mt < 4; ++mt)
#pragma unroll
      for (int nt = 0; nt < 2; ++nt)
        acc[mt][nt] = wmma_bf16(a0[mt], b0[nt], acc[mt][nt]);
#pragma unroll
    for (int mt = 0; mt < 4; ++mt) a0[mt] = a1[mt];
#pragma unroll
    for (int nt = 0; nt < 2; ++nt) b0[nt] = b1[nt];
  }
#pragma unroll
  for (int mt = 0; mt < 4; ++mt)
#pragma unroll
    for (int nt = 0; nt < 2; ++nt)
      acc[mt][nt] = wmma_bf16(a0[mt], b0[nt], acc[mt][nt]);

#pragma unroll
  for (int mt = 0; mt < 4; ++mt)
#pragma unroll
    for (int nt = 0; nt < 2; ++nt) {
      const int col = rn + nt * 16 + l16;
      const float bv = bias[col];
#pragma unroll
      for (int v = 0; v < 8; ++v) {
        const int row = rm + mt * 16 + v + hi * 8;
        const float val = acc[mt][nt][v] + bv;
        const size_t flat = (size_t)row * N + col;
        if (MODE == 0) {
          ((unsigned short*)Cout)[flat] = f2bf_u16(val);
        } else if (MODE == 1) {
          const size_t tr = (flat & ~(size_t)0xFFFF) | ((flat & 63) << 10) |
                            ((flat >> 6) & 1023);
          ((unsigned short*)Cout)[tr] = f2bf_u16(val);
        } else {
          ((float*)Cout)[flat] = val;
        }
      }
    }
}

// ---------------------------------------------------------------------------
// Fused attention: one workgroup per (bh, 64-row block).
// LDS: 64x1024 f32 score slab (256KB) + reduction scratch.
// Mask is applied during the softmax max pass (contiguous uint4 mask loads).
// ---------------------------------------------------------------------------
__global__ __launch_bounds__(256) void attention_fused(
    const unsigned short* __restrict__ Q,    // [128][1024][64] bf16
    const unsigned short* __restrict__ Kb,   // [128][1024][64] bf16
    const unsigned short* __restrict__ Vt,   // [128][64][1024] bf16
    const unsigned char* __restrict__ mask,  // [8][1024][1024] bool
    float* __restrict__ attn_out,            // [128][1024][1024] f32
    unsigned short* __restrict__ ctx) {      // [128][1024][64] bf16
  extern __shared__ float smem[];
  float* S    = smem;           // 65536 floats
  float* red  = smem + 65536;   // 256 floats
  float* invr = red + 256;      // 64 floats

  const int tid  = threadIdx.x;
  const int wave = tid >> 5;
  const int lane = tid & 31;
  const int l16  = lane & 15;
  const int hi   = lane >> 4;
  const int bh   = blockIdx.y;
  const int i0   = blockIdx.x * 64;
  const size_t base = (size_t)bh << 16;   // bh * 1024 * 64

  // Preload this wave's Q fragments: 4 m-tiles x 2 k-steps (DPH=64).
  v16bf qf[4][2];
#pragma unroll
  for (int mt = 0; mt < 4; ++mt)
#pragma unroll
    for (int ks = 0; ks < 2; ++ks)
      qf[mt][ks] = load_frag_a(Q + base + (size_t)(i0 + mt * 16 + l16) * 64 +
                               ks * 32 + hi * 8);

  v8f zero = {0.f, 0.f, 0.f, 0.f, 0.f, 0.f, 0.f, 0.f};

  // Phase A: S = scale * Q K^T into LDS; wave owns 8 key tiles (pipelined).
  {
    const unsigned short* Kw = Kb + base + (size_t)l16 * 64 + hi * 8;
    v16bf kb0[2];
    kb0[0] = load_frag_b(Kw + (size_t)(wave * 8 + 0) * 16 * 64);
    kb0[1] = load_frag_b(Kw + (size_t)(wave * 8 + 0) * 16 * 64 + 32);
    for (int t8 = 0; t8 < 8; ++t8) {
      const int jc = (wave * 8 + t8) * 16 + l16;
      v16bf kb1[2];
      if (t8 < 7) {
        const unsigned short* pn = Kw + (size_t)(wave * 8 + t8 + 1) * 16 * 64;
        kb1[0] = load_frag_b(pn);
        kb1[1] = load_frag_b(pn + 32);
      }
      v8f acc[4];
#pragma unroll
      for (int mt = 0; mt < 4; ++mt) acc[mt] = zero;
#pragma unroll
      for (int ks = 0; ks < 2; ++ks)
#pragma unroll
        for (int mt = 0; mt < 4; ++mt)
          acc[mt] = wmma_bf16(qf[mt][ks], kb0[ks], acc[mt]);
#pragma unroll
      for (int mt = 0; mt < 4; ++mt)
#pragma unroll
        for (int v = 0; v < 8; ++v) {
          const int rl = mt * 16 + v + hi * 8;
          S[rl * 1024 + jc] = acc[mt][v] * 0.125f;   // DPH^-0.5
        }
      kb0[0] = kb1[0];
      kb0[1] = kb1[1];
    }
  }
  __syncthreads();

  // Phase B: mask + softmax over rows (4 threads per row, 256 cols each).
  const int r = tid >> 2, sub = tid & 3;
  float* Sr = S + r * 1024 + sub * 256;
  {
    const unsigned char* mr = mask + ((size_t)(bh & 7) << 20) +
                              (size_t)(i0 + r) * 1024 + sub * 256;
    float mx = -3.4e38f;
    for (int c = 0; c < 256; c += 16) {
      const uint4 mb = *reinterpret_cast<const uint4*>(mr + c);
      unsigned w[4] = {mb.x, mb.y, mb.z, mb.w};
#pragma unroll
      for (int q = 0; q < 4; ++q) {
        float4 sv = *reinterpret_cast<float4*>(Sr + c + q * 4);
        float* f = &sv.x;
#pragma unroll
        for (int b = 0; b < 4; ++b) {
          if ((w[q] >> (8 * b)) & 0xFFu) f[b] = -1e9f;   // mask==True -> -1e9
          mx = fmaxf(mx, f[b]);
        }
        *reinterpret_cast<float4*>(Sr + c + q * 4) = sv;
      }
    }
    red[(r << 2) + sub] = mx;
  }
  __syncthreads();
  const float rowmax = fmaxf(fmaxf(red[r << 2], red[(r << 2) + 1]),
                             fmaxf(red[(r << 2) + 2], red[(r << 2) + 3]));
  __syncthreads();
  {
    float ssum = 0.f;
    for (int c = 0; c < 256; c += 4) {
      float4 sv = *reinterpret_cast<float4*>(Sr + c);
      sv.x = __expf(sv.x - rowmax);
      sv.y = __expf(sv.y - rowmax);
      sv.z = __expf(sv.z - rowmax);
      sv.w = __expf(sv.w - rowmax);
      ssum += sv.x + sv.y + sv.z + sv.w;
      *reinterpret_cast<float4*>(Sr + c) = sv;
    }
    red[(r << 2) + sub] = ssum;
  }
  __syncthreads();
  if (sub == 0)
    invr[r] = 1.0f / (red[r << 2] + red[(r << 2) + 1] + red[(r << 2) + 2] +
                      red[(r << 2) + 3]);
  __syncthreads();

  // Phase C: normalize in LDS + stream attention to HBM (float4, coalesced).
  {
    float* aout = attn_out + ((size_t)bh << 20) + (size_t)i0 * 1024;
    for (int n = tid * 4; n < 65536; n += 1024) {
      const float iv = invr[n >> 10];
      float4 sv = *reinterpret_cast<float4*>(S + n);
      sv.x *= iv; sv.y *= iv; sv.z *= iv; sv.w *= iv;
      *reinterpret_cast<float4*>(S + n) = sv;
      *reinterpret_cast<float4*>(aout + n) = sv;
    }
  }
  __syncthreads();

  // Phase D: ctx = P @ V. Wave owns 1 m-tile x 2 dph-tiles, K=1024 (pipelined).
  const int mtile = wave >> 1;
  const int ng    = wave & 1;
  v8f cacc[2];
  cacc[0] = zero; cacc[1] = zero;
  const int prow = mtile * 16 + l16;
  const unsigned short* Vbase[2];
#pragma unroll
  for (int nt = 0; nt < 2; ++nt)
    Vbase[nt] = Vt + base + (size_t)((ng * 2 + nt) * 16 + l16) * 1024 + hi * 8;

  v16bf vb0[2];
  vb0[0] = load_frag_b(Vbase[0]);
  vb0[1] = load_frag_b(Vbase[1]);
  for (int kk = 0; kk < 1024; kk += 32) {
    v16bf vb1[2];
    if (kk < 992) {
      vb1[0] = load_frag_b(Vbase[0] + kk + 32);
      vb1[1] = load_frag_b(Vbase[1] + kk + 32);
    }
    const float* ps = S + prow * 1024 + kk + hi * 8;
    v16bf af;
#pragma unroll
    for (int j = 0; j < 8; ++j) af[j] = f2bf(ps[j]);
#pragma unroll
    for (int j = 0; j < 8; ++j) af[8 + j] = f2bf(ps[16 + j]);
#pragma unroll
    for (int nt = 0; nt < 2; ++nt) cacc[nt] = wmma_bf16(af, vb0[nt], cacc[nt]);
    vb0[0] = vb1[0];
    vb0[1] = vb1[1];
  }
#pragma unroll
  for (int nt = 0; nt < 2; ++nt) {
    const int dph = (ng * 2 + nt) * 16 + l16;
#pragma unroll
    for (int v = 0; v < 8; ++v) {
      const int i = i0 + mtile * 16 + v + hi * 8;
      ctx[base + (size_t)i * 64 + dph] = f2bf_u16(cacc[nt][v]);
    }
  }
}

// ---------------------------------------------------------------------------
__global__ __launch_bounds__(256) void residual_layernorm(
    const float* __restrict__ resid, const float* __restrict__ proj,
    const float* __restrict__ gamma, const float* __restrict__ beta,
    float* __restrict__ out) {
  __shared__ float s1[256], s2[256];
  const int t = threadIdx.x;
  const size_t rbase = (size_t)blockIdx.x * 1024;
  float x[4], sum = 0.f, sq = 0.f;
#pragma unroll
  for (int j = 0; j < 4; ++j) {
    const int c = t + j * 256;
    const float v = resid[rbase + c] + proj[rbase + c];
    x[j] = v;
    sum += v;
    sq += v * v;
  }
  s1[t] = sum;
  s2[t] = sq;
  __syncthreads();
  for (int o = 128; o > 0; o >>= 1) {
    if (t < o) { s1[t] += s1[t + o]; s2[t] += s2[t + o]; }
    __syncthreads();
  }
  const float mean = s1[0] * (1.f / 1024.f);
  const float var  = s2[0] * (1.f / 1024.f) - mean * mean;
  const float rs   = rsqrtf(var + 1e-5f);
#pragma unroll
  for (int j = 0; j < 4; ++j) {
    const int c = t + j * 256;
    out[rbase + c] = (x[j] - mean) * rs * gamma[c] + beta[c];
  }
}

// ---------------------------------------------------------------------------
extern "C" void kernel_launch(void* const* d_in, const int* in_sizes, int n_in,
                              void* d_out, int out_size, void* d_ws,
                              size_t ws_size, hipStream_t stream) {
  (void)in_sizes; (void)n_in; (void)out_size; (void)ws_size;
  const float* key_in = (const float*)d_in[0];
  const float* val_in = (const float*)d_in[1];
  const float* qry_in = (const float*)d_in[2];
  const unsigned char* mask = (const unsigned char*)d_in[3];  // jnp bool (1B)
  const float* Wq = (const float*)d_in[4];
  const float* bq = (const float*)d_in[5];
  const float* Wk = (const float*)d_in[6];
  const float* bk = (const float*)d_in[7];
  const float* Wv = (const float*)d_in[8];
  const float* bv = (const float*)d_in[9];
  const float* Wo = (const float*)d_in[10];
  const float* bo = (const float*)d_in[11];
  const float* gamma = (const float*)d_in[12];
  const float* beta  = (const float*)d_in[13];

  float* out  = (float*)d_out;                  // [8,1024,1024]
  float* attn = out + (size_t)8 * 1024 * 1024;  // [128,1024,1024]

  char* ws = (char*)d_ws;
  const size_t MB = 1024 * 1024;
  unsigned short* qb    = (unsigned short*)(ws + 0 * MB);
  unsigned short* kb    = (unsigned short*)(ws + 16 * MB);
  unsigned short* vb    = (unsigned short*)(ws + 32 * MB);
  unsigned short* WqT   = (unsigned short*)(ws + 48 * MB);
  unsigned short* WkT   = (unsigned short*)(ws + 50 * MB);
  unsigned short* WvT   = (unsigned short*)(ws + 52 * MB);
  unsigned short* WoT   = (unsigned short*)(ws + 54 * MB);
  unsigned short* qact  = (unsigned short*)(ws + 56 * MB);
  unsigned short* kact  = (unsigned short*)(ws + 72 * MB);
  unsigned short* vactT = (unsigned short*)(ws + 88 * MB);
  unsigned short* ctx   = (unsigned short*)(ws + 104 * MB);
  float*          oprj  = (float*)(ws + 120 * MB);

  const int NACT = 8 * 1024 * 1024;
  cvt_bf16<<<NACT / 256, 256, 0, stream>>>(qry_in, qb, NACT);
  cvt_bf16<<<NACT / 256, 256, 0, stream>>>(key_in, kb, NACT);
  cvt_bf16<<<NACT / 256, 256, 0, stream>>>(val_in, vb, NACT);
  dim3 tg(16, 16);
  transpose_cvt_1024<<<tg, 256, 0, stream>>>(Wq, WqT);
  transpose_cvt_1024<<<tg, 256, 0, stream>>>(Wk, WkT);
  transpose_cvt_1024<<<tg, 256, 0, stream>>>(Wv, WvT);
  transpose_cvt_1024<<<tg, 256, 0, stream>>>(Wo, WoT);

  dim3 gg(1024 / 128, 8192 / 128);
  gemm_bf16_wmma<0><<<gg, 256, 0, stream>>>(qb, WqT, bq, qact, 8192, 1024, 1024);
  gemm_bf16_wmma<0><<<gg, 256, 0, stream>>>(kb, WkT, bk, kact, 8192, 1024, 1024);
  gemm_bf16_wmma<1><<<gg, 256, 0, stream>>>(vb, WvT, bv, vactT, 8192, 1024, 1024);

  const size_t smem_bytes = (65536 + 256 + 64) * sizeof(float);  // ~257KB
  attention_fused<<<dim3(16, 128), 256, smem_bytes, stream>>>(qact, kact, vactT,
                                                              mask, attn, ctx);

  gemm_bf16_wmma<2><<<gg, 256, 0, stream>>>(ctx, WoT, bo, oprj, 8192, 1024, 1024);

  residual_layernorm<<<8192, 256, 0, stream>>>(qry_in, oprj, gamma, beta, out);
}
